// CIE_48052094108098
// MI455X (gfx1250) — matrix operand order repaired and checked by reading, compiled, not run on hardware
//
#include <hip/hip_runtime.h>
#include <math.h>

// ---------------------------------------------------------------------------
// Shapes (fixed by the reference)
// ---------------------------------------------------------------------------
#define S_LEN 2048
#define BATCH 4
#define DMODEL 512
#define NHH 4          // heads per module
#define DHEAD 128
#define FFDIM 2048
#define HIDG 128       // gater hidden
#define QKVN 1536

typedef __bf16 bf16;
typedef __attribute__((ext_vector_type(16))) __bf16 v16bf;
typedef __attribute__((ext_vector_type(8)))  __bf16 v8bf;
typedef __attribute__((ext_vector_type(8)))  float  v8f;

static __device__ inline v8f v8f_zero() {
    v8f r;
#pragma unroll
    for (int i = 0; i < 8; ++i) r[i] = 0.0f;
    return r;
}

// ---------------------------------------------------------------------------
// fp32 -> bf16 cast (weights)
// ---------------------------------------------------------------------------
__global__ void cie_cvt_bf16(const float* __restrict__ src, bf16* __restrict__ dst, int n) {
    int i = blockIdx.x * blockDim.x + threadIdx.x;
    if (i < n) dst[i] = (bf16)src[i];
}

// ---------------------------------------------------------------------------
// Half-sequence means: fs[b,d] = mean(seq[0:1024]), fb = mean(seq[1024:2048])
// ---------------------------------------------------------------------------
__global__ void cie_halfmean(const float* __restrict__ seq,
                             float* __restrict__ fs, float* __restrict__ fb) {
    int idx = blockIdx.x * blockDim.x + threadIdx.x;   // 0..B*D-1, idx = b*512+d
    if (idx >= BATCH * DMODEL) return;
    int b = idx >> 9, d = idx & 511;
    float s0 = 0.f, s1 = 0.f;
    for (int t = 0; t < 1024; ++t) {
        s0 += seq[((size_t)t * BATCH + b) * DMODEL + d];
        s1 += seq[((size_t)(t + 1024) * BATCH + b) * DMODEL + d];
    }
    fs[idx] = s0 * (1.0f / 1024.0f);
    fb[idx] = s1 * (1.0f / 1024.0f);
}

// ---------------------------------------------------------------------------
// Gater MLP: gates[b,{0,1}] = sigmoid(relu([fs,fb,|fs-fb|] @ w1^T + b1) @ w2^T + b2)
// ---------------------------------------------------------------------------
__global__ __launch_bounds__(HIDG) void cie_gater(
    const float* __restrict__ fs, const float* __restrict__ fb,
    const float* __restrict__ w1, const float* __restrict__ b1,
    const float* __restrict__ w2, const float* __restrict__ b2,
    float* __restrict__ gates) {
    __shared__ float hh[HIDG];
    int t = threadIdx.x;
    for (int b = 0; b < BATCH; ++b) {
        const float* fsb = fs + b * DMODEL;
        const float* fbb = fb + b * DMODEL;
        const float* wr  = w1 + (size_t)t * (3 * DMODEL);
        float acc = b1[t];
        for (int k = 0; k < DMODEL; ++k) {
            float a = fsb[k], c = fbb[k];
            acc += a * wr[k] + c * wr[DMODEL + k] + fabsf(a - c) * wr[2 * DMODEL + k];
        }
        hh[t] = fmaxf(acc, 0.0f);
        __syncthreads();
        if (t < 2) {
            float o = b2[t];
            for (int j = 0; j < HIDG; ++j) o += hh[j] * w2[t * HIDG + j];
            gates[b * 2 + t] = 1.0f / (1.0f + expf(-o));
        }
        __syncthreads();
    }
}

// ---------------------------------------------------------------------------
// LayerNorm over D, one block per token row; writes bf16 xn
// ---------------------------------------------------------------------------
__global__ __launch_bounds__(256) void cie_layernorm(
    const float* __restrict__ x, const float* __restrict__ g,
    const float* __restrict__ bta, bf16* __restrict__ xn) {
    __shared__ float red[16];
    const int row = blockIdx.x;                 // row = s*B + b
    const int tid = threadIdx.x;
    const float* xr = x + (size_t)row * DMODEL;
    float a0 = xr[tid], a1 = xr[tid + 256];
    float s = a0 + a1, sq = a0 * a0 + a1 * a1;
#pragma unroll
    for (int off = 16; off >= 1; off >>= 1) {
        s  += __shfl_xor(s, off, 32);
        sq += __shfl_xor(sq, off, 32);
    }
    if ((tid & 31) == 0) { red[tid >> 5] = s; red[8 + (tid >> 5)] = sq; }
    __syncthreads();
    float ts = 0.f, tq = 0.f;
#pragma unroll
    for (int i = 0; i < 8; ++i) { ts += red[i]; tq += red[8 + i]; }
    float mu  = ts * (1.0f / DMODEL);
    float var = tq * (1.0f / DMODEL) - mu * mu;
    float inv = rsqrtf(var + 1e-5f);
    bf16* orow = xn + (size_t)row * DMODEL;
    orow[tid]       = (bf16)((a0 - mu) * inv * g[tid] + bta[tid]);
    orow[tid + 256] = (bf16)((a1 - mu) * inv * g[tid + 256] + bta[tid + 256]);
}

// ---------------------------------------------------------------------------
// Generic bf16 WMMA GEMM:  out(M,N) = A(M,K) @ W(N,K)^T + bias [+src] [GELU]
// Block = 256 thr = 8 waves; block tile 128(M)x128(N); wave tile 64x32.
// flags: 1=GELU  2=add src(f32)  4=write f32  8=write bf16
// ---------------------------------------------------------------------------
__global__ __launch_bounds__(256) void cie_gemm_bf16(
    const bf16* __restrict__ A, const bf16* __restrict__ W,
    const float* __restrict__ bias, const float* __restrict__ addsrc,
    float* __restrict__ outf, bf16* __restrict__ outb,
    int M, int N, int K, int flags) {
    const int lane = threadIdx.x & 31;
    const int wave = threadIdx.x >> 5;
    const int hf   = lane >> 4;       // which 16-lane half
    const int l15  = lane & 15;
    const int wm = wave & 1;          // 2 M-groups
    const int wn = wave >> 1;         // 4 N-groups
    const int row0 = blockIdx.y * 128 + wm * 64;
    const int col0 = blockIdx.x * 128 + wn * 32;

    v8f acc[4][2];
#pragma unroll
    for (int i = 0; i < 4; ++i)
#pragma unroll
        for (int j = 0; j < 2; ++j) acc[i][j] = v8f_zero();

    for (int k = 0; k < K; k += 32) {
        v16bf af[4];
#pragma unroll
        for (int mt = 0; mt < 4; ++mt) {
            const bf16* ap = A + (size_t)(row0 + mt * 16 + l15) * K + k;
            v8bf lo = *(const v8bf*)(ap + hf * 16);         // K = k + hf*16 .. +7
            v8bf hi = *(const v8bf*)(ap + hf * 16 + 8);     // K = k+hf*16+8 .. +7
#pragma unroll
            for (int i = 0; i < 8; ++i) { af[mt][i] = lo[i]; af[mt][i + 8] = hi[i]; }
        }
        v16bf wf[2];
#pragma unroll
        for (int nt = 0; nt < 2; ++nt) {
            const bf16* wp = W + (size_t)(col0 + nt * 16 + l15) * K + k + hf * 16;
            wf[nt] = *(const v16bf*)wp;                     // 16 contiguous K values
        }
#pragma unroll
        for (int mt = 0; mt < 4; ++mt)
#pragma unroll
            for (int nt = 0; nt < 2; ++nt)
                acc[mt][nt] = __builtin_amdgcn_wmma_f32_16x16x32_bf16(
                    false, af[mt], false, wf[nt], (short)0, acc[mt][nt], false, false);
    }

#pragma unroll
    for (int mt = 0; mt < 4; ++mt) {
#pragma unroll
        for (int nt = 0; nt < 2; ++nt) {
            const int col = col0 + nt * 16 + l15;
            const float bv = bias[col];
#pragma unroll
            for (int r = 0; r < 8; ++r) {
                const int row = row0 + mt * 16 + hf * 8 + r;
                float v = acc[mt][nt][r] + bv;
                if (flags & 1) v = 0.5f * v * (1.0f + erff(v * 0.70710678118654752f));
                size_t idx = (size_t)row * N + col;
                if (flags & 2) v += addsrc[idx];
                if (flags & 4) outf[idx] = v;
                if (flags & 8) outb[idx] = (bf16)v;
            }
        }
    }
}

// ---------------------------------------------------------------------------
// Gated V transpose: vt[b,h,d,s] = qkv_v[s,b,h,d] * gate[b,module]  (bf16)
// ---------------------------------------------------------------------------
__global__ void cie_vgate_tr(const bf16* __restrict__ qkv,
                             const float* __restrict__ gates, int module,
                             bf16* __restrict__ vt) {
    size_t idx = (size_t)blockIdx.x * blockDim.x + threadIdx.x;
    if (idx >= (size_t)BATCH * NHH * DHEAD * S_LEN) return;
    int s = (int)(idx & (S_LEN - 1));
    int d = (int)((idx >> 11) & (DHEAD - 1));
    int h = (int)((idx >> 18) & (NHH - 1));
    int b = (int)(idx >> 20);
    float g = gates[b * 2 + module];
    float v = (float)qkv[((size_t)s * BATCH + b) * QKVN + 2 * DMODEL + h * DHEAD + d];
    vt[idx] = (bf16)(v * g);
}

// ---------------------------------------------------------------------------
// Flash attention over one fully-allowed key half (block mask => no masking).
// One wave per block; 16 q rows; streams 32-key chunks; online softmax.
// grid = (64 q-tiles, 2 q-halves, B*HH); module selects same/opposite half.
// ---------------------------------------------------------------------------
__global__ __launch_bounds__(32) void cie_attn(
    const bf16* __restrict__ qkv, const bf16* __restrict__ vt,
    bf16* __restrict__ out, int module) {
    __shared__ alignas(16) bf16 lds_p[16 * 32];
    const int lane = threadIdx.x;
    const int hf   = lane >> 4;
    const int l15  = lane & 15;
    const int qhalf = blockIdx.y;
    const int bh = blockIdx.z;
    const int b = bh >> 2, h = bh & 3;
    const int qbase = qhalf * 1024 + blockIdx.x * 16;
    const int khalf = (module == 0) ? qhalf : (1 - qhalf);
    const int kbase0 = khalf * 1024;
    const float scale = 0.088388347648318447f;   // 1/sqrt(128)

    // Q fragments (A-operand), loop invariant: 16 rows x 128 dh
    v16bf aq[4];
    {
        const bf16* qp = qkv + ((size_t)(qbase + l15) * BATCH + b) * QKVN + h * DHEAD;
#pragma unroll
        for (int c = 0; c < 4; ++c) {
            v8bf lo = *(const v8bf*)(qp + c * 32 + hf * 16);
            v8bf hi = *(const v8bf*)(qp + c * 32 + hf * 16 + 8);
#pragma unroll
            for (int i = 0; i < 8; ++i) { aq[c][i] = lo[i]; aq[c][i + 8] = hi[i]; }
        }
    }

    v8f oacc[8];
#pragma unroll
    for (int j = 0; j < 8; ++j) oacc[j] = v8f_zero();
    float mrun[8], lrun[8];
#pragma unroll
    for (int r = 0; r < 8; ++r) { mrun[r] = -1e30f; lrun[r] = 0.0f; }

    for (int kc = 0; kc < 32; ++kc) {
        const int kb = kbase0 + kc * 32;
        v8f s0 = v8f_zero(), s1 = v8f_zero();
#pragma unroll
        for (int c = 0; c < 4; ++c) {
            const bf16* kp0 = qkv + ((size_t)(kb + l15) * BATCH + b) * QKVN
                              + DMODEL + h * DHEAD + c * 32 + hf * 16;
            const bf16* kp1 = qkv + ((size_t)(kb + 16 + l15) * BATCH + b) * QKVN
                              + DMODEL + h * DHEAD + c * 32 + hf * 16;
            v16bf bk0 = *(const v16bf*)kp0;
            v16bf bk1 = *(const v16bf*)kp1;
            s0 = __builtin_amdgcn_wmma_f32_16x16x32_bf16(false, aq[c], false, bk0,
                                                         (short)0, s0, false, false);
            s1 = __builtin_amdgcn_wmma_f32_16x16x32_bf16(false, aq[c], false, bk1,
                                                         (short)0, s1, false, false);
        }
        // Online softmax per q-row (row m = r + 8*hf lives on 16 lanes of one half)
#pragma unroll
        for (int r = 0; r < 8; ++r) {
            float a  = s0[r] * scale;
            float bb = s1[r] * scale;
            float x = fmaxf(a, bb);
#pragma unroll
            for (int off = 8; off >= 1; off >>= 1) x = fmaxf(x, __shfl_xor(x, off, 16));
            float mnew = fmaxf(mrun[r], x);
            float fsc  = expf(mrun[r] - mnew);
            float p0 = expf(a - mnew);
            float p1 = expf(bb - mnew);
            float ps = p0 + p1;
#pragma unroll
            for (int off = 8; off >= 1; off >>= 1) ps += __shfl_xor(ps, off, 16);
            lrun[r] = lrun[r] * fsc + ps;
            mrun[r] = mnew;
#pragma unroll
            for (int j = 0; j < 8; ++j) oacc[j][r] *= fsc;
            int mr = r + hf * 8;
            lds_p[mr * 32 + l15]      = (bf16)p0;
            lds_p[mr * 32 + 16 + l15] = (bf16)p1;
        }
        __syncthreads();
        // Re-load P as A-operand (16x32)
        v16bf pa;
        {
            v8bf lo = *(const v8bf*)(&lds_p[l15 * 32 + hf * 16]);
            v8bf hi = *(const v8bf*)(&lds_p[l15 * 32 + hf * 16 + 8]);
#pragma unroll
            for (int i = 0; i < 8; ++i) { pa[i] = lo[i]; pa[i + 8] = hi[i]; }
        }
        __syncthreads();
        // O += P @ V ; V^T layout makes B-operand contiguous
#pragma unroll
        for (int j = 0; j < 8; ++j) {
            const bf16* vp = vt + (((size_t)(b * NHH + h) * DHEAD) + j * 16 + l15) * S_LEN
                             + kb + hf * 16;
            v16bf bv = *(const v16bf*)vp;
            oacc[j] = __builtin_amdgcn_wmma_f32_16x16x32_bf16(false, pa, false, bv,
                                                              (short)0, oacc[j], false, false);
        }
    }

    // Finalize: O /= l, store bf16 (S,B,D)
#pragma unroll
    for (int r = 0; r < 8; ++r) {
        float inv = 1.0f / lrun[r];
        int row = qbase + r + hf * 8;
#pragma unroll
        for (int j = 0; j < 8; ++j) {
            float v = oacc[j][r] * inv;
            out[((size_t)row * BATCH + b) * DMODEL + h * DHEAD + j * 16 + l15] = (bf16)v;
        }
    }
}

// ---------------------------------------------------------------------------
// Host launcher
// ---------------------------------------------------------------------------
extern "C" void kernel_launch(void* const* d_in, const int* in_sizes, int n_in,
                              void* d_out, int out_size, void* d_ws, size_t ws_size,
                              hipStream_t stream) {
    (void)in_sizes; (void)n_in; (void)out_size; (void)ws_size;
    const float* seq     = (const float*)d_in[0];
    const float* g_w1    = (const float*)d_in[1];
    const float* g_b1    = (const float*)d_in[2];
    const float* g_w2    = (const float*)d_in[3];
    const float* g_b2    = (const float*)d_in[4];
    const float* ln_g    = (const float*)d_in[5];
    const float* ln_b    = (const float*)d_in[6];
    const float* hm_in_w = (const float*)d_in[7];
    const float* hm_in_b = (const float*)d_in[8];
    const float* hm_ou_w = (const float*)d_in[9];
    const float* hm_ou_b = (const float*)d_in[10];
    const float* ht_in_w = (const float*)d_in[11];
    const float* ht_in_b = (const float*)d_in[12];
    const float* ht_ou_w = (const float*)d_in[13];
    const float* ht_ou_b = (const float*)d_in[14];
    const float* f_w1    = (const float*)d_in[15];
    const float* f_b1    = (const float*)d_in[16];
    const float* f_w2    = (const float*)d_in[17];
    const float* f_b2    = (const float*)d_in[18];
    float* outp = (float*)d_out;

    // ---- workspace carve-up (bf16 regions first; all offsets pow2-aligned) ----
    char* ws = (char*)d_ws;
    size_t off = 0;
    auto abf = [&](size_t elems) { bf16* p = (bf16*)(ws + off); off += elems * sizeof(bf16); return p; };
    bf16* w_hi = abf((size_t)QKVN * DMODEL);      // homo_in_w  bf16
    bf16* w_ti = abf((size_t)QKVN * DMODEL);      // het_in_w
    bf16* w_ho = abf((size_t)DMODEL * DMODEL);    // homo_out_w
    bf16* w_to = abf((size_t)DMODEL * DMODEL);    // het_out_w
    bf16* w_f1 = abf((size_t)FFDIM * DMODEL);     // ffn_w1
    bf16* w_f2 = abf((size_t)DMODEL * FFDIM);     // ffn_w2
    bf16* xn   = abf((size_t)S_LEN * BATCH * DMODEL);
    bf16* qkvh = abf((size_t)S_LEN * BATCH * QKVN);
    bf16* qkvt = abf((size_t)S_LEN * BATCH * QKVN);
    bf16* vth  = abf((size_t)BATCH * NHH * DHEAD * S_LEN);
    bf16* vtt  = abf((size_t)BATCH * NHH * DHEAD * S_LEN);
    bf16* ath  = abf((size_t)S_LEN * BATCH * DMODEL);
    bf16* att  = abf((size_t)S_LEN * BATCH * DMODEL);
    bf16* ybf  = abf((size_t)S_LEN * BATCH * DMODEL);
    bf16* hbuf = abf((size_t)S_LEN * BATCH * FFDIM);
    auto af32 = [&](size_t elems) { float* p = (float*)(ws + off); off += elems * sizeof(float); return p; };
    float* yf    = af32((size_t)S_LEN * BATCH * DMODEL);
    float* fs    = af32((size_t)BATCH * DMODEL);
    float* fb    = af32((size_t)BATCH * DMODEL);
    float* gates = af32(8);

    // ---- 1) weight casts ----
    auto cvt = [&](const float* s, bf16* d, int n) {
        cie_cvt_bf16<<<(n + 255) / 256, 256, 0, stream>>>(s, d, n);
    };
    cvt(hm_in_w, w_hi, QKVN * DMODEL);
    cvt(ht_in_w, w_ti, QKVN * DMODEL);
    cvt(hm_ou_w, w_ho, DMODEL * DMODEL);
    cvt(ht_ou_w, w_to, DMODEL * DMODEL);
    cvt(f_w1,    w_f1, FFDIM * DMODEL);
    cvt(f_w2,    w_f2, DMODEL * FFDIM);

    // ---- 2) gater ----
    cie_halfmean<<<(BATCH * DMODEL + 255) / 256, 256, 0, stream>>>(seq, fs, fb);
    cie_gater<<<1, HIDG, 0, stream>>>(fs, fb, g_w1, g_b1, g_w2, g_b2, gates);

    // ---- 3) layernorm ----
    cie_layernorm<<<S_LEN * BATCH, 256, 0, stream>>>(seq, ln_g, ln_b, xn);

    const int M = S_LEN * BATCH;     // 8192
    // ---- 4) QKV projections (write bf16) ----
    cie_gemm_bf16<<<dim3(QKVN / 128, M / 128), 256, 0, stream>>>(
        xn, w_hi, hm_in_b, nullptr, nullptr, qkvh, M, QKVN, DMODEL, /*bf16*/ 8);
    cie_gemm_bf16<<<dim3(QKVN / 128, M / 128), 256, 0, stream>>>(
        xn, w_ti, ht_in_b, nullptr, nullptr, qkvt, M, QKVN, DMODEL, 8);

    // ---- 5) gated V transpose ----
    const int VTN = BATCH * NHH * DHEAD * S_LEN;
    cie_vgate_tr<<<(VTN + 255) / 256, 256, 0, stream>>>(qkvh, gates, 0, vth);
    cie_vgate_tr<<<(VTN + 255) / 256, 256, 0, stream>>>(qkvt, gates, 1, vtt);

    // ---- 6) flash attention (homo: same half; het: opposite half) ----
    cie_attn<<<dim3(64, 2, BATCH * NHH), 32, 0, stream>>>(qkvh, vth, ath, 0);
    cie_attn<<<dim3(64, 2, BATCH * NHH), 32, 0, stream>>>(qkvt, vtt, att, 1);

    // ---- 7) output projections with residual: y = seq + homo + het ----
    cie_gemm_bf16<<<dim3(DMODEL / 128, M / 128), 256, 0, stream>>>(
        ath, w_ho, hm_ou_b, seq, yf, nullptr, M, DMODEL, DMODEL, /*add|f32*/ 2 | 4);
    cie_gemm_bf16<<<dim3(DMODEL / 128, M / 128), 256, 0, stream>>>(
        att, w_to, ht_ou_b, yf, yf, ybf, M, DMODEL, DMODEL, /*add|f32|bf16*/ 2 | 4 | 8);

    // ---- 8) FFN: h = gelu(y@w1^T+b1); out = y + h@w2^T+b2 ----
    cie_gemm_bf16<<<dim3(FFDIM / 128, M / 128), 256, 0, stream>>>(
        ybf, w_f1, f_b1, nullptr, nullptr, hbuf, M, FFDIM, DMODEL, /*gelu|bf16*/ 1 | 8);
    cie_gemm_bf16<<<dim3(DMODEL / 128, M / 128), 256, 0, stream>>>(
        hbuf, w_f2, f_b2, yf, outp, nullptr, M, DMODEL, FFDIM, /*add|f32*/ 2 | 4);
}
